// CoSSM_77206332113398
// MI455X (gfx1250) — matrix operand order repaired and checked by reading, not compiled, not run on hardware
//
#include <hip/hip_runtime.h>
#include <hip/hip_bf16.h>
#include <math.h>

// ---------------- types ----------------
typedef __attribute__((ext_vector_type(16))) __bf16 v16bf;
typedef __attribute__((ext_vector_type(8)))  __bf16 v8bf;
typedef __attribute__((ext_vector_type(8)))  float  v8f;

#define DEVINL static __device__ __forceinline__

// ---------------- problem constants ----------------
constexpr int Bt     = 4;
constexpr int Lt     = 512;
constexpr int DIN    = 256;
constexpr int DMODEL = 1024;
constexpr int DINNER = 2048;
constexpr int DSTATE = 16;
constexpr int DTRANK = 64;
constexpr int NTOK   = Bt * Lt;             // 2048 tokens
constexpr int XZW    = 2 * DINNER;          // 4096
constexpr int DBLW   = DTRANK + 2 * DSTATE; // 96

// float -> bf16 round-to-nearest-even
DEVINL __bf16 f2bf(float f) {
  union { float f; unsigned u; } v; v.f = f;
  unsigned r = v.u + 0x7FFFu + ((v.u >> 16) & 1u);
  unsigned short h = (unsigned short)(r >> 16);
  union { unsigned short s; __bf16 b; } o; o.s = h;
  return o.b;
}

DEVINL float siluf(float x) { return x / (1.f + __expf(-x)); }

DEVINL v16bf load_afrag(const __bf16* p) {       // two 16B chunks: K {0..7} and {16..23} (per lane-half)
  const v8bf a0 = *(const v8bf*)(p);
  const v8bf a1 = *(const v8bf*)(p + 16);
  v16bf a;
#pragma unroll
  for (int i = 0; i < 8; ++i) { a[i] = a0[i]; a[8 + i] = a1[i]; }
  return a;
}

DEVINL v8f wmma_bf16(v16bf a, v16bf b, v8f c) {
  return __builtin_amdgcn_wmma_f32_16x16x32_bf16(
      /*neg_a=*/false, a, /*neg_b=*/false, b,
      /*c_mod=*/(short)0, c, /*reuse_a=*/false, /*reuse_b=*/false);
}

// ---------------- bf16 WMMA GEMM:  C[M,N] = A[M,K] * B[N,K]^T ----------------
// Each wave computes a 32x32 macro-tile (2x2 of 16x16) -> 4 WMMA per 4 fragment loads.
// Block = 4 waves in 2x2 -> 64x64 per block. Requires: M % 64 == 0, N % 32 == 0, K % 32 == 0.
__global__ void __launch_bounds__(128)
wmma_gemm_bf16(const __bf16* __restrict__ A, const __bf16* __restrict__ Bm,
               float* __restrict__ C, int M, int N, int K)
{
  const int lane = threadIdx.x;                  // 0..31
  const int wave = threadIdx.y;                  // 0..3
  const int wy   = wave >> 1;                    // M direction within block
  const int wx   = wave & 1;                     // N direction within block
  const int m0   = (blockIdx.y * 2 + wy) * 32;   // two M tiles: m0, m0+16
  const int n0   = (blockIdx.x * 2 + wx) * 32;   // two N tiles: n0, n0+16
  if (n0 >= N) return;                           // whole wave exits together (N % 32 == 0)

  const int r  = lane & 15;
  const int hs = lane >> 4;

  const __bf16* aRow0 = A + (size_t)(m0 + r) * K + hs * 8;
  const __bf16* aRow1 = aRow0 + (size_t)16 * K;
  const __bf16* bRow0 = Bm + (size_t)(n0 + r) * K + hs * 8;
  const __bf16* bRow1 = bRow0 + (size_t)16 * K;

  v8f acc00 = {0.f,0.f,0.f,0.f,0.f,0.f,0.f,0.f};
  v8f acc01 = acc00, acc10 = acc00, acc11 = acc00;

  for (int kk = 0; kk < K; kk += 32) {
    v16bf a0 = load_afrag(aRow0 + kk);
    v16bf a1 = load_afrag(aRow1 + kk);
    v16bf b0 = load_afrag(bRow0 + kk);
    v16bf b1 = load_afrag(bRow1 + kk);

    if (kk + 32 < K) {                           // gfx1250 global_prefetch_b8 of next K-slab
      __builtin_prefetch(aRow0 + kk + 32, 0, 1);
      __builtin_prefetch(aRow1 + kk + 32, 0, 1);
      __builtin_prefetch(bRow0 + kk + 32, 0, 1);
      __builtin_prefetch(bRow1 + kk + 32, 0, 1);
    }

    acc00 = wmma_bf16(a0, b0, acc00);
    acc01 = wmma_bf16(a0, b1, acc01);
    acc10 = wmma_bf16(a1, b0, acc10);
    acc11 = wmma_bf16(a1, b1, acc11);
  }

  // D layout: VGPR i, lane l -> row = i + 8*hs, col = lane&15
  {
    float* c00 = C + (size_t)(m0 + hs * 8) * N + n0 + r;
    float* c10 = c00 + (size_t)16 * N;
#pragma unroll
    for (int i = 0; i < 8; ++i) {
      c00[(size_t)i * N]      = acc00[i];
      c00[(size_t)i * N + 16] = acc01[i];
      c10[(size_t)i * N]      = acc10[i];
      c10[(size_t)i * N + 16] = acc11[i];
    }
  }
}

// ---------------- residual conv branch ----------------
// out[bl,c] = relu(conv3(x)/sqrt(1+eps)*bg + bb) + conv1(x)
__global__ void __launch_bounds__(256)
resbranch_kernel(const float* __restrict__ x, const float* __restrict__ cw,
                 const float* __restrict__ bg, const float* __restrict__ bbv,
                 const float* __restrict__ sw, float* __restrict__ out)
{
  __shared__ float xs[3 * DIN];
  const int bl = blockIdx.y;
  const int b  = bl >> 9;
  const int l  = bl & (Lt - 1);
  const int t  = threadIdx.x;
  const float* xb = x + (size_t)(b * Lt) * DIN;
  xs[t]           = (l > 0)      ? xb[(size_t)(l - 1) * DIN + t] : 0.f;
  xs[DIN + t]     =                xb[(size_t)l       * DIN + t];
  xs[2 * DIN + t] = (l < Lt - 1) ? xb[(size_t)(l + 1) * DIN + t] : 0.f;
  __syncthreads();

  const int c = blockIdx.x * 256 + t;
  const float* w = cw + (size_t)c * (DIN * 3);
  const float* s = sw + (size_t)c * DIN;
  float acc = 0.f, skip = 0.f;
  for (int i = 0; i < DIN; ++i) {
    float xm = xs[i], x0 = xs[DIN + i], xp = xs[2 * DIN + i];
    acc  = fmaf(w[i * 3], xm, acc);
    acc  = fmaf(w[i * 3 + 1], x0, acc);
    acc  = fmaf(w[i * 3 + 2], xp, acc);
    skip = fmaf(s[i], x0, skip);
  }
  const float scale = rsqrtf(1.f + 1e-5f);
  float y = fmaf(acc * scale, bg[c], bbv[c]);
  y = fmaxf(y, 0.f);
  out[(size_t)bl * DMODEL + c] = y + skip;
}

// ---------------- elementwise casts ----------------
__global__ void cast_bf16_kernel(const float* __restrict__ s, __bf16* __restrict__ d, int n)
{
  int i = blockIdx.x * 256 + threadIdx.x;
  if (i < n) d[i] = f2bf(s[i]);
}

// ---------------- causal depthwise conv (K=4) + SiLU; optional sequence reversal ----------------
__global__ void __launch_bounds__(256)
dwconv_silu_kernel(const float* __restrict__ xz, const float* __restrict__ w,
                   const float* __restrict__ bias, float* __restrict__ u32,
                   __bf16* __restrict__ ubf, int reverse)
{
  int idx = blockIdx.x * 256 + threadIdx.x;       // NTOK * DINNER
  int d  = idx & (DINNER - 1);
  int bl = idx >> 11;
  int b  = bl >> 9;
  int l  = bl & (Lt - 1);
  float acc = bias[d];
#pragma unroll
  for (int k = 0; k < 4; ++k) {
    int t = l - 3 + k;
    if (t >= 0) {
      int sl = reverse ? (Lt - 1 - t) : t;        // xb = xi[:, ::-1]
      acc = fmaf(w[d * 4 + k], xz[((size_t)(b * Lt + sl)) * XZW + d], acc);
    }
  }
  float s = siluf(acc);
  u32[idx] = s;
  ubf[idx] = f2bf(s);
}

// ---------------- dt slice (first 64 of 96 cols) to bf16 ----------------
__global__ void extract_dt_kernel(const float* __restrict__ dbl, __bf16* __restrict__ dt)
{
  int i = blockIdx.x * 256 + threadIdx.x;         // NTOK * 64
  int row = i >> 6, col = i & 63;
  dt[i] = f2bf(dbl[row * DBLW + col]);
}

// ---------------- selective scan: one thread per (b, d), 16-state in registers ----------------
__global__ void __launch_bounds__(256)
scan_kernel(const float* __restrict__ delta_raw, const float* __restrict__ dt_bias,
            const float* __restrict__ A_log, const float* __restrict__ Dp,
            const float* __restrict__ u, const float* __restrict__ dbl,
            float* __restrict__ ys)
{
  int gid = blockIdx.x * 256 + threadIdx.x;       // Bt * DINNER = 8192
  int b = gid >> 11;
  int d = gid & (DINNER - 1);

  float Ar[DSTATE], h[DSTATE];
#pragma unroll
  for (int n = 0; n < DSTATE; ++n) { Ar[n] = -__expf(A_log[d * DSTATE + n]); h[n] = 0.f; }
  const float biasv = dt_bias[d];
  const float Dd    = Dp[d];

  for (int l = 0; l < Lt; ++l) {
    size_t tok = (size_t)b * Lt + l;
    float dr = delta_raw[tok * DINNER + d] + biasv;
    float delta = (dr > 20.f) ? dr : log1pf(__expf(dr));   // softplus
    float uu = u[tok * DINNER + d];
    const float* bc = dbl + tok * DBLW;                    // B at +64, C at +80 (uniform per block)
    float du = delta * uu;
    float y = 0.f;
#pragma unroll
    for (int n = 0; n < DSTATE; ++n) {
      float dA = __expf(delta * Ar[n]);
      h[n] = fmaf(dA, h[n], du * bc[DTRANK + n]);
      y = fmaf(h[n], bc[DTRANK + DSTATE + n], y);
    }
    ys[tok * DINNER + d] = fmaf(uu, Dd, y);
  }
}

// ---------------- (yf + reverse(yb)) * silu(z) -> bf16 ----------------
__global__ void __launch_bounds__(256)
combine_gate_kernel(const float* __restrict__ ysf, const float* __restrict__ ysb,
                    const float* __restrict__ xz, __bf16* __restrict__ yg)
{
  int idx = blockIdx.x * 256 + threadIdx.x;       // NTOK * DINNER
  int e  = idx & (DINNER - 1);
  int bl = idx >> 11;
  int b  = bl >> 9, l = bl & (Lt - 1);
  float z = xz[(size_t)bl * XZW + DINNER + e];
  float y = ysf[idx] + ysb[((size_t)(b * Lt + (Lt - 1 - l))) * DINNER + e];
  yg[idx] = f2bf(y * siluf(z));
}

// ---------------- LayerNorm + residual ----------------
__global__ void __launch_bounds__(256)
ln_res_kernel(const float* __restrict__ xin, const float* __restrict__ branch,
              const float* __restrict__ g, const float* __restrict__ bb,
              float* __restrict__ out)
{
  __shared__ float s1[256], s2[256];
  const int row = blockIdx.x, tid = threadIdx.x;
  const float* xr = xin + (size_t)row * DMODEL;
  float a = 0.f, q = 0.f;
#pragma unroll
  for (int k = 0; k < 4; ++k) { float v = xr[tid + k * 256]; a += v; q = fmaf(v, v, q); }
  s1[tid] = a; s2[tid] = q;
  __syncthreads();
  for (int st = 128; st > 0; st >>= 1) {
    if (tid < st) { s1[tid] += s1[tid + st]; s2[tid] += s2[tid + st]; }
    __syncthreads();
  }
  float mean = s1[0] * (1.f / DMODEL);
  float var  = s2[0] * (1.f / DMODEL) - mean * mean;
  float rstd = rsqrtf(var + 1e-6f);
  const float* br = branch + (size_t)row * DMODEL;
  float* o = out + (size_t)row * DMODEL;
#pragma unroll
  for (int k = 0; k < 4; ++k) {
    int c = tid + k * 256;
    o[c] = br[c] + (xr[c] - mean) * rstd * g[c] + bb[c];
  }
}

// =======================================================================
extern "C" void kernel_launch(void* const* d_in, const int* in_sizes, int n_in,
                              void* d_out, int out_size, void* d_ws, size_t ws_size,
                              hipStream_t stream)
{
  (void)in_sizes; (void)n_in; (void)out_size; (void)ws_size;

  const float* g_x       = (const float*)d_in[0];
  const float* r_x       = (const float*)d_in[1];
  const float* e_conv_w  = (const float*)d_in[2];
  const float* e_bn_g    = (const float*)d_in[3];
  const float* e_bn_b    = (const float*)d_in[4];
  const float* e_skip_w  = (const float*)d_in[5];
  const float* g_conv_w  = (const float*)d_in[6];
  const float* g_bn_g    = (const float*)d_in[7];
  const float* g_bn_b    = (const float*)d_in[8];
  const float* g_skip_w  = (const float*)d_in[9];
  const float* in_proj_w = (const float*)d_in[10];
  const float* conv_w_f  = (const float*)d_in[11];
  const float* conv_b_f  = (const float*)d_in[12];
  const float* x_proj_f  = (const float*)d_in[13];
  const float* dt_w_f    = (const float*)d_in[14];
  const float* dt_bias_f = (const float*)d_in[15];
  const float* A_log_f   = (const float*)d_in[16];
  const float* D_f       = (const float*)d_in[17];
  const float* conv_w_b  = (const float*)d_in[18];
  const float* conv_b_b  = (const float*)d_in[19];
  const float* x_proj_b  = (const float*)d_in[20];
  const float* dt_w_b    = (const float*)d_in[21];
  const float* dt_bias_b = (const float*)d_in[22];
  const float* A_log_b   = (const float*)d_in[23];
  const float* D_b       = (const float*)d_in[24];
  const float* out_proj_w= (const float*)d_in[25];
  const float* ln1_g     = (const float*)d_in[26];
  const float* ln1_b     = (const float*)d_in[27];
  const float* ln2_g     = (const float*)d_in[28];
  const float* ln2_b     = (const float*)d_in[29];

  // ---------------- workspace carve-up ----------------
  char* base = (char*)d_ws;
  size_t off = 0;
  auto alloc = [&](size_t bytes) -> char* {
    off = (off + 255) & ~(size_t)255;
    char* p = base + off;
    off += bytes;
    return p;
  };
  float*  g1      = (float*)alloc((size_t)NTOK * DMODEL * 4);
  float*  r1      = (float*)alloc((size_t)NTOK * DMODEL * 4);
  __bf16* Wip     = (__bf16*)alloc((size_t)XZW * DMODEL * 2);     // in_proj (4096x1024)
  __bf16* Wop     = (__bf16*)alloc((size_t)DMODEL * DINNER * 2);  // out_proj (1024x2048)
  __bf16* Wxpf    = (__bf16*)alloc((size_t)DBLW * DINNER * 2);
  __bf16* Wxpb    = (__bf16*)alloc((size_t)DBLW * DINNER * 2);
  __bf16* Wdtf    = (__bf16*)alloc((size_t)DINNER * DTRANK * 2);
  __bf16* Wdtb    = (__bf16*)alloc((size_t)DINNER * DTRANK * 2);
  __bf16* xA      = (__bf16*)alloc((size_t)NTOK * DMODEL * 2);
  float*  xz      = (float*)alloc((size_t)NTOK * XZW * 4);
  float*  u32f    = (float*)alloc((size_t)NTOK * DINNER * 4);
  float*  u32b    = (float*)alloc((size_t)NTOK * DINNER * 4);
  __bf16* ubff    = (__bf16*)alloc((size_t)NTOK * DINNER * 2);
  __bf16* ubfb    = (__bf16*)alloc((size_t)NTOK * DINNER * 2);
  float*  dblf    = (float*)alloc((size_t)NTOK * DBLW * 4);
  float*  dblb    = (float*)alloc((size_t)NTOK * DBLW * 4);
  __bf16* dtbf_f  = (__bf16*)alloc((size_t)NTOK * DTRANK * 2);
  __bf16* dtbf_b  = (__bf16*)alloc((size_t)NTOK * DTRANK * 2);
  float*  deltaf  = (float*)alloc((size_t)NTOK * DINNER * 4);
  float*  deltab  = (float*)alloc((size_t)NTOK * DINNER * 4);
  float*  ysf     = (float*)alloc((size_t)NTOK * DINNER * 4);
  float*  ysb     = (float*)alloc((size_t)NTOK * DINNER * 4);
  __bf16* ygate   = (__bf16*)alloc((size_t)NTOK * DINNER * 2);
  float*  mout    = (float*)alloc((size_t)NTOK * DMODEL * 4);

  const dim3 blk256(256);
  const dim3 gemmBlk(32, 4);
  auto ceilDiv = [](int a, int b) { return (a + b - 1) / b; };
  // grid for 64x64 macro-blocks: x over N (ceil to 64), y over M (M % 64 == 0)
  auto gemmGrid = [&](int Mv, int Nv) { return dim3(ceilDiv(Nv, 64), Mv / 64); };

  // ---------------- conv branches ----------------
  {
    dim3 grid(DMODEL / 256, NTOK);
    resbranch_kernel<<<grid, blk256, 0, stream>>>(g_x, e_conv_w, e_bn_g, e_bn_b, e_skip_w, g1);
    resbranch_kernel<<<grid, blk256, 0, stream>>>(r_x, g_conv_w, g_bn_g, g_bn_b, g_skip_w, r1);
  }

  // ---------------- weight casts (bf16) ----------------
  cast_bf16_kernel<<<ceilDiv(XZW * DMODEL, 256), blk256, 0, stream>>>(in_proj_w, Wip, XZW * DMODEL);
  cast_bf16_kernel<<<ceilDiv(DMODEL * DINNER, 256), blk256, 0, stream>>>(out_proj_w, Wop, DMODEL * DINNER);
  cast_bf16_kernel<<<ceilDiv(DBLW * DINNER, 256), blk256, 0, stream>>>(x_proj_f, Wxpf, DBLW * DINNER);
  cast_bf16_kernel<<<ceilDiv(DBLW * DINNER, 256), blk256, 0, stream>>>(x_proj_b, Wxpb, DBLW * DINNER);
  cast_bf16_kernel<<<ceilDiv(DINNER * DTRANK, 256), blk256, 0, stream>>>(dt_w_f, Wdtf, DINNER * DTRANK);
  cast_bf16_kernel<<<ceilDiv(DINNER * DTRANK, 256), blk256, 0, stream>>>(dt_w_b, Wdtb, DINNER * DTRANK);

  // ---------------- two BiMamba passes ----------------
  for (int pass = 0; pass < 2; ++pass) {
    const float* branch = pass ? r1 : g1;
    const float* lng = pass ? ln2_g : ln1_g;
    const float* lnb = pass ? ln2_b : ln1_b;
    float* outp = (float*)d_out + (size_t)pass * NTOK * DMODEL;

    // branch -> bf16
    cast_bf16_kernel<<<ceilDiv(NTOK * DMODEL, 256), blk256, 0, stream>>>(branch, xA, NTOK * DMODEL);

    // in_proj: (2048x1024)·(4096x1024)^T -> xz
    wmma_gemm_bf16<<<gemmGrid(NTOK, XZW), gemmBlk, 0, stream>>>(xA, Wip, xz, NTOK, XZW, DMODEL);

    // depthwise conv + silu, fwd + reversed
    dwconv_silu_kernel<<<ceilDiv(NTOK * DINNER, 256), blk256, 0, stream>>>(xz, conv_w_f, conv_b_f, u32f, ubff, 0);
    dwconv_silu_kernel<<<ceilDiv(NTOK * DINNER, 256), blk256, 0, stream>>>(xz, conv_w_b, conv_b_b, u32b, ubfb, 1);

    // x_proj: (2048x2048)·(96x2048)^T -> dbl
    wmma_gemm_bf16<<<gemmGrid(NTOK, DBLW), gemmBlk, 0, stream>>>(ubff, Wxpf, dblf, NTOK, DBLW, DINNER);
    wmma_gemm_bf16<<<gemmGrid(NTOK, DBLW), gemmBlk, 0, stream>>>(ubfb, Wxpb, dblb, NTOK, DBLW, DINNER);

    // dt slice -> bf16
    extract_dt_kernel<<<ceilDiv(NTOK * DTRANK, 256), blk256, 0, stream>>>(dblf, dtbf_f);
    extract_dt_kernel<<<ceilDiv(NTOK * DTRANK, 256), blk256, 0, stream>>>(dblb, dtbf_b);

    // delta: (2048x64)·(2048x64)^T -> delta_raw
    wmma_gemm_bf16<<<gemmGrid(NTOK, DINNER), gemmBlk, 0, stream>>>(dtbf_f, Wdtf, deltaf, NTOK, DINNER, DTRANK);
    wmma_gemm_bf16<<<gemmGrid(NTOK, DINNER), gemmBlk, 0, stream>>>(dtbf_b, Wdtb, deltab, NTOK, DINNER, DTRANK);

    // selective scan (softplus + dA/dBu fused)
    scan_kernel<<<dim3(Bt * DINNER / 256), blk256, 0, stream>>>(deltaf, dt_bias_f, A_log_f, D_f, u32f, dblf, ysf);
    scan_kernel<<<dim3(Bt * DINNER / 256), blk256, 0, stream>>>(deltab, dt_bias_b, A_log_b, D_b, u32b, dblb, ysb);

    // (yf + rev(yb)) * silu(z) -> bf16
    combine_gate_kernel<<<ceilDiv(NTOK * DINNER, 256), blk256, 0, stream>>>(ysf, ysb, xz, ygate);

    // out_proj: (2048x2048)·(1024x2048)^T -> mamba_out
    wmma_gemm_bf16<<<gemmGrid(NTOK, DMODEL), gemmBlk, 0, stream>>>(ygate, Wop, mout, NTOK, DMODEL, DINNER);

    // LayerNorm + residual -> d_out
    ln_res_kernel<<<dim3(NTOK), blk256, 0, stream>>>(mout, branch, lng, lnb, outp);
  }
}
